// SimpleRNN_76544907149659
// MI455X (gfx1250) — matrix operand order repaired and checked by reading, not compiled, not run on hardware
//
#include <hip/hip_runtime.h>
#include <hip/hip_bf16.h>
#include <math.h>

#define INPUT_  512
#define HIDDEN_ 1024
#define OUTPUT_ 512
#define B_      64
#define T_      1024
#define BT_     (B_ * T_)

typedef __attribute__((ext_vector_type(16))) __bf16 v16bf;
typedef __attribute__((ext_vector_type(8)))  __bf16 v8bf;
typedef __attribute__((ext_vector_type(8)))  float  v8f;

// ---------------------------------------------------------------------------
// Fragment loaders, this code's contiguous K order (same for A and B, not AMD's table order):
// A 16x32 bf16: lane m holds K = kb..kb+15 contiguous, kb = 16*(lane>=16).
// B 32x16 bf16: lane holds one column of B (= row of the row-major weight
//               matrix), K = kb..kb+15 contiguous, kb = 16*(lane>=16).
// ---------------------------------------------------------------------------
__device__ __forceinline__ v16bf load_fragA(const __bf16* base, int ld, int lane) {
    int r  = lane & 15;
    int kb = (lane >> 4) << 4;
    const __bf16* p = base + (size_t)r * ld + kb;
    v8bf lo = *(const v8bf*)(p);
    v8bf hi = *(const v8bf*)(p + 8);
    v16bf f;
#pragma unroll
    for (int i = 0; i < 8; ++i) { f[i] = lo[i]; f[i + 8] = hi[i]; }
    return f;
}

__device__ __forceinline__ v16bf load_fragB(const __bf16* base, int ld, int lane) {
    int n  = lane & 15;
    int kb = (lane >> 4) << 4;
    const __bf16* p = base + (size_t)n * ld + kb;
    v8bf lo = *(const v8bf*)(p);
    v8bf hi = *(const v8bf*)(p + 8);
    v16bf f;
#pragma unroll
    for (int i = 0; i < 8; ++i) { f[i] = lo[i]; f[i + 8] = hi[i]; }
    return f;
}

// B fragment from LDS (row stride = HIDDEN_), lowers to ds_load_b128 pairs.
__device__ __forceinline__ v16bf load_fragB_lds(const __bf16* base, int lane) {
    int n  = lane & 15;
    int kb = (lane >> 4) << 4;
    const __bf16* p = base + n * HIDDEN_ + kb;
    v8bf lo = *(const v8bf*)(p);
    v8bf hi = *(const v8bf*)(p + 8);
    v16bf f;
#pragma unroll
    for (int i = 0; i < 8; ++i) { f[i] = lo[i]; f[i + 8] = hi[i]; }
    return f;
}

__device__ __forceinline__ v8f wmma_bf16(v16bf a, v16bf b, v8f c) {
    return __builtin_amdgcn_wmma_f32_16x16x32_bf16(false, a, false, b, (short)0, c, false, false);
}

// ---------------------------------------------------------------------------
__global__ void cast_f32_bf16(const float* __restrict__ src, __bf16* __restrict__ dst, long n) {
    long i      = (long)blockIdx.x * blockDim.x + threadIdx.x;
    long stride = (long)gridDim.x * blockDim.x;
    for (; i < n; i += stride) dst[i] = (__bf16)src[i];
}

// ---------------------------------------------------------------------------
// Projection. M indexed tb = t*B + b so each 16-row tile is one timestep.
// Output written in C-fragment layout: xp[((t*4+tm)*64+tn)*256 + lane*8 + v].
// Ping-pong (unroll-by-2) software pipeline: no register-rotation copies.
// ---------------------------------------------------------------------------
__global__ void __launch_bounds__(256) proj_kernel(
        const __bf16* __restrict__ xbf, const __bf16* __restrict__ wih,
        const float* __restrict__ bih, const float* __restrict__ bhh,
        float* __restrict__ xp) {
    int lane = threadIdx.x & 31;
    int w    = blockIdx.x * (blockDim.x >> 5) + (threadIdx.x >> 5);
    int ng   = w & 15;                 // 16 N-groups of 64 columns (H=1024)
    int mt   = w >> 4;                 // 4096 tb-tiles
    if (mt >= BT_ / 16) return;
    int t  = mt >> 2;                  // timestep
    int tm = mt & 3;                   // batch tile (b = tm*16 + r)
    const int K32 = INPUT_ / 32;       // 16 (even)
    const int ldA = T_ * INPUT_;       // row stride between consecutive b at fixed t
    const __bf16* abase = xbf + ((size_t)(tm * 16) * T_ + t) * INPUT_;
    const __bf16* wb[4];
#pragma unroll
    for (int j = 0; j < 4; ++j) wb[j] = wih + (size_t)(ng * 64 + j * 16) * INPUT_;

    v8f zero = {0.f, 0.f, 0.f, 0.f, 0.f, 0.f, 0.f, 0.f};
    v8f acc[4] = {zero, zero, zero, zero};

    // stage 0 prologue
    v16bf a0 = load_fragA(abase, ldA, lane);
    v16bf b0[4], b1[4];
    v16bf a1;
#pragma unroll
    for (int j = 0; j < 4; ++j) b0[j] = load_fragB(wb[j], INPUT_, lane);

    for (int kk = 0; kk < K32 - 2; kk += 2) {
        a1 = load_fragA(abase + (kk + 1) * 32, ldA, lane);
#pragma unroll
        for (int j = 0; j < 4; ++j) b1[j] = load_fragB(wb[j] + (kk + 1) * 32, INPUT_, lane);
#pragma unroll
        for (int j = 0; j < 4; ++j) acc[j] = wmma_bf16(a0, b0[j], acc[j]);

        a0 = load_fragA(abase + (kk + 2) * 32, ldA, lane);
#pragma unroll
        for (int j = 0; j < 4; ++j) b0[j] = load_fragB(wb[j] + (kk + 2) * 32, INPUT_, lane);
#pragma unroll
        for (int j = 0; j < 4; ++j) acc[j] = wmma_bf16(a1, b1[j], acc[j]);
    }
    // epilogue: kk = K32-2, K32-1
    a1 = load_fragA(abase + (K32 - 1) * 32, ldA, lane);
#pragma unroll
    for (int j = 0; j < 4; ++j) b1[j] = load_fragB(wb[j] + (K32 - 1) * 32, INPUT_, lane);
#pragma unroll
    for (int j = 0; j < 4; ++j) acc[j] = wmma_bf16(a0, b0[j], acc[j]);
#pragma unroll
    for (int j = 0; j < 4; ++j) acc[j] = wmma_bf16(a1, b1[j], acc[j]);

    int N = lane & 15;
#pragma unroll
    for (int j = 0; j < 4; ++j) {
        int   tn   = ng * 4 + j;
        int   h    = tn * 16 + N;
        float bias = bih[h] + bhh[h];
        v8f   r    = acc[j] + bias;               // same column for all 8 rows
        v8f*  ptile = (v8f*)(xp + ((((size_t)t * 4 + tm) * 64) + tn) * 256);
        ptile[lane] = r;
    }
}

// ---------------------------------------------------------------------------
// Persistent recurrence: hs[t] = tanh(xp[t] + hs[t-1] @ W_hh^T)
// 32 blocks x 8 waves = the 4x64 tiles of [64,1024]. W_hh rows staged in LDS
// once (64KB), reused for all 1024 steps. Ping-pong pipeline on the h-state
// A-fragments; B-fragments come from LDS one stage ahead of their WMMA.
// ---------------------------------------------------------------------------
__global__ void __launch_bounds__(256) rnn_kernel(
        const float* __restrict__ xp, const __bf16* __restrict__ whh,
        __bf16* __restrict__ hs, int* __restrict__ syncc) {
    __shared__ __bf16 lwhh[32 * HIDDEN_];          // 64 KB of this block's W_hh rows

    int tid  = threadIdx.x;
    int lane = tid & 31;
    int widx = tid >> 5;                           // 0..7
    int w    = blockIdx.x * 8 + widx;              // 0..255
    int tm   = w & 3;                              // batch tile
    int tn   = w >> 2;                             // hidden tile (0..63)
    int tn_l = widx >> 2;                          // 0..1: local row-group in LDS
    int Mb   = tm * 16, Nb = tn * 16;
    int N    = lane & 15;
    int moff = (lane >> 4) * 8;
    int nblocks = (int)gridDim.x;
    const int K32 = HIDDEN_ / 32;                  // 32 (even)

    // Stage W_hh rows [blockIdx.x*32 .. +31] into LDS (once, reused 1024x).
    {
        const uint4* src = (const uint4*)(whh + (size_t)blockIdx.x * 32 * HIDDEN_);
        uint4*       dst = (uint4*)lwhh;
        for (int i = tid; i < 32 * HIDDEN_ / 8; i += 256) dst[i] = src[i];
    }
    __syncthreads();

    const __bf16* lB = lwhh + tn_l * 16 * HIDDEN_;

    for (int t = 0; t < T_; ++t) {
        const v8f* xtile = (const v8f*)(xp + ((((size_t)t * 4 + tm) * 64) + tn) * 256);
        if (t + 1 < T_)
            __builtin_prefetch((const float*)(xp + ((((size_t)(t + 1) * 4 + tm) * 64) + tn) * 256) +
                               lane * 8, 0, 1);

        v8f acc = xtile[lane];                      // one 32B vector load

        if (t > 0) {
            const __bf16* hp = hs + (size_t)(t - 1) * B_ * HIDDEN_ + (size_t)Mb * HIDDEN_;
            v16bf a0 = load_fragA(hp, HIDDEN_, lane);
            v16bf b0 = load_fragB_lds(lB, lane);
            v16bf a1, b1;
            for (int kk = 0; kk < K32 - 2; kk += 2) {
                a1  = load_fragA(hp + (kk + 1) * 32, HIDDEN_, lane);
                b1  = load_fragB_lds(lB + (kk + 1) * 32, lane);
                acc = wmma_bf16(a0, b0, acc);
                a0  = load_fragA(hp + (kk + 2) * 32, HIDDEN_, lane);
                b0  = load_fragB_lds(lB + (kk + 2) * 32, lane);
                acc = wmma_bf16(a1, b1, acc);
            }
            a1  = load_fragA(hp + (K32 - 1) * 32, HIDDEN_, lane);
            b1  = load_fragB_lds(lB + (K32 - 1) * 32, lane);
            acc = wmma_bf16(a0, b0, acc);
            acc = wmma_bf16(a1, b1, acc);
        }

        __bf16* ht = hs + (size_t)t * B_ * HIDDEN_;
#pragma unroll
        for (int v = 0; v < 8; ++v)
            ht[(size_t)(Mb + moff + v) * HIDDEN_ + Nb + N] = (__bf16)tanhf(acc[v]);

        if (t < T_ - 1) {                           // device-wide split barrier
            __threadfence();
            __syncthreads();
            if (tid == 0) {
                __hip_atomic_fetch_add(syncc, 1, __ATOMIC_ACQ_REL, __HIP_MEMORY_SCOPE_AGENT);
                int target = (t + 1) * nblocks;
                while (__hip_atomic_load(syncc, __ATOMIC_ACQUIRE, __HIP_MEMORY_SCOPE_AGENT) < target)
                    __builtin_amdgcn_s_sleep(2);
            }
            __syncthreads();
        }
    }
}

// ---------------------------------------------------------------------------
// Output head: out[b][t][o] = sum_h hs[t][b][h]*W_out[o][h] + b_out[o]
// M = rows of hs[T*B, H] (tb = t*B + b); ping-pong pipeline like proj.
// ---------------------------------------------------------------------------
__global__ void __launch_bounds__(256) head_kernel(
        const __bf16* __restrict__ hs, const __bf16* __restrict__ wout,
        const float* __restrict__ bout, float* __restrict__ out) {
    int lane = threadIdx.x & 31;
    int w    = blockIdx.x * (blockDim.x >> 5) + (threadIdx.x >> 5);
    int ng   = w & 7;                  // 8 N-groups of 64 (O=512)
    int mt   = w >> 3;                 // 4096 M tiles
    if (mt >= BT_ / 16) return;
    const int K32 = HIDDEN_ / 32;      // 32 (even)

    v8f zero = {0.f, 0.f, 0.f, 0.f, 0.f, 0.f, 0.f, 0.f};
    v8f acc[4] = {zero, zero, zero, zero};

    const __bf16* abase = hs + (size_t)mt * 16 * HIDDEN_;
    const __bf16* wb[4];
#pragma unroll
    for (int j = 0; j < 4; ++j) wb[j] = wout + (size_t)(ng * 64 + j * 16) * HIDDEN_;

    v16bf a0 = load_fragA(abase, HIDDEN_, lane);
    v16bf b0[4], b1[4];
    v16bf a1;
#pragma unroll
    for (int j = 0; j < 4; ++j) b0[j] = load_fragB(wb[j], HIDDEN_, lane);

    for (int kk = 0; kk < K32 - 2; kk += 2) {
        a1 = load_fragA(abase + (kk + 1) * 32, HIDDEN_, lane);
#pragma unroll
        for (int j = 0; j < 4; ++j) b1[j] = load_fragB(wb[j] + (kk + 1) * 32, HIDDEN_, lane);
#pragma unroll
        for (int j = 0; j < 4; ++j) acc[j] = wmma_bf16(a0, b0[j], acc[j]);

        a0 = load_fragA(abase + (kk + 2) * 32, HIDDEN_, lane);
#pragma unroll
        for (int j = 0; j < 4; ++j) b0[j] = load_fragB(wb[j] + (kk + 2) * 32, HIDDEN_, lane);
#pragma unroll
        for (int j = 0; j < 4; ++j) acc[j] = wmma_bf16(a1, b1[j], acc[j]);
    }
    a1 = load_fragA(abase + (K32 - 1) * 32, HIDDEN_, lane);
#pragma unroll
    for (int j = 0; j < 4; ++j) b1[j] = load_fragB(wb[j] + (K32 - 1) * 32, HIDDEN_, lane);
#pragma unroll
    for (int j = 0; j < 4; ++j) acc[j] = wmma_bf16(a0, b0[j], acc[j]);
#pragma unroll
    for (int j = 0; j < 4; ++j) acc[j] = wmma_bf16(a1, b1[j], acc[j]);

    int N    = lane & 15;
    int moff = (lane >> 4) * 8;
#pragma unroll
    for (int j = 0; j < 4; ++j) {
        int   o  = ng * 64 + j * 16 + N;
        float bo = bout[o];
#pragma unroll
        for (int v = 0; v < 8; ++v) {
            int tb = mt * 16 + moff + v;           // tb = t*B + b
            int t  = tb >> 6;
            int b  = tb & (B_ - 1);
            out[((size_t)b * T_ + t) * OUTPUT_ + o] = acc[j][v] + bo;
        }
    }
}

// ---------------------------------------------------------------------------
extern "C" void kernel_launch(void* const* d_in, const int* in_sizes, int n_in,
                              void* d_out, int out_size, void* d_ws, size_t ws_size,
                              hipStream_t stream) {
    (void)in_sizes; (void)n_in; (void)out_size; (void)ws_size;
    const float* x    = (const float*)d_in[0];
    const float* Wih  = (const float*)d_in[1];
    const float* Whh  = (const float*)d_in[2];
    const float* bih  = (const float*)d_in[3];
    const float* bhh  = (const float*)d_in[4];
    const float* Wout = (const float*)d_in[5];
    const float* bout = (const float*)d_in[6];
    float* out = (float*)d_out;

    char*  ws  = (char*)d_ws;
    size_t off = 0;
    auto wsalloc = [&](size_t bytes) -> char* {
        char* p = ws + off;
        off += (bytes + 255) & ~(size_t)255;
        return p;
    };
    __bf16* xbf   = (__bf16*)wsalloc((size_t)BT_ * INPUT_ * 2);        //  64 MB
    __bf16* wih   = (__bf16*)wsalloc((size_t)HIDDEN_ * INPUT_ * 2);    //   1 MB
    __bf16* whh   = (__bf16*)wsalloc((size_t)HIDDEN_ * HIDDEN_ * 2);   //   2 MB
    __bf16* wout  = (__bf16*)wsalloc((size_t)OUTPUT_ * HIDDEN_ * 2);   //   1 MB
    float*  xp    = (float*) wsalloc((size_t)T_ * B_ * HIDDEN_ * 4);   // 256 MB (frag-tiled)
    __bf16* hs    = (__bf16*)wsalloc((size_t)T_ * B_ * HIDDEN_ * 2);   // 128 MB [T,B,H]
    int*    syncc = (int*)   wsalloc(256);

    hipMemsetAsync(syncc, 0, 4, stream);   // reset step-barrier counter (capture-safe)

    cast_f32_bf16<<<2048, 256, 0, stream>>>(x,    xbf,  (long)BT_ * INPUT_);
    cast_f32_bf16<<<512,  256, 0, stream>>>(Wih,  wih,  (long)HIDDEN_ * INPUT_);
    cast_f32_bf16<<<512,  256, 0, stream>>>(Whh,  whh,  (long)HIDDEN_ * HIDDEN_);
    cast_f32_bf16<<<512,  256, 0, stream>>>(Wout, wout, (long)OUTPUT_ * HIDDEN_);

    proj_kernel<<<8192, 256, 0, stream>>>(xbf, wih, bih, bhh, xp);
    rnn_kernel<<<32, 256, 0, stream>>>(xp, whh, hs, syncc);
    head_kernel<<<4096, 256, 0, stream>>>(hs, wout, bout, out);
}
